// DocSenModel_64166811402942
// MI455X (gfx1250) — matrix operand order repaired and checked by reading, not compiled, not run on hardware
//
#include <hip/hip_runtime.h>
#include <math.h>

typedef __attribute__((ext_vector_type(16))) _Float16 v16h;
typedef __attribute__((ext_vector_type(8)))  _Float16 v8h;
typedef __attribute__((ext_vector_type(8)))  float    v8f;

constexpr int kH    = 50;          // hidden
constexpr int kE    = 300;         // embedding dim
constexpr int kKpad = 320;         // K padded to multiple of 32
constexpr int kS    = 64;          // sentences
constexpr int kW    = 64;          // words / sentence
constexpr int kM    = kS * kW;     // 4096 GEMM rows
constexpr int kNpad = 64;          // N padded 50 -> 64
constexpr int kO    = 5;           // classes

// ---------------------------------------------------------------------------
// Kernel 0a: gather + convert A.  Apack[m][k] = f16(emb[doc[m]][k]), k<300,
// zero for 300<=k<320.  One block per row m; thread t packs halves 2t,2t+1.
// ---------------------------------------------------------------------------
__global__ __launch_bounds__(160) void pack_A_kernel(
    const int* __restrict__ doc, const float* __restrict__ emb,
    unsigned* __restrict__ Apack /* viewed as uint, kKpad/2 per row */)
{
    const int m   = blockIdx.x;
    const int tid = threadIdx.x;                  // 0..159
    const float* row = emb + (long)doc[m] * kE;

    union { _Float16 h[2]; unsigned u; } p;
    if (tid < kE / 2) {                           // tid < 150: both halves valid
        p.h[0] = (_Float16)row[2 * tid];
        p.h[1] = (_Float16)row[2 * tid + 1];
    } else {
        p.u = 0u;                                 // K padding 300..319
    }
    Apack[m * (kKpad / 2) + tid] = p.u;
}

// ---------------------------------------------------------------------------
// Kernel 0b: convert/pad B.  Bpack[n][k] = f16(W_word[n][k]) for n<50,k<300,
// else 0.  64 x 320 halves.
// ---------------------------------------------------------------------------
__global__ __launch_bounds__(256) void pack_B_kernel(
    const float* __restrict__ W_word, _Float16* __restrict__ Bpack)
{
    const int idx = blockIdx.x * blockDim.x + threadIdx.x; // 0 .. 64*320-1
    if (idx >= kNpad * kKpad) return;
    const int n = idx / kKpad;
    const int k = idx % kKpad;
    const float v = (n < kH && k < kE) ? W_word[n * kE + k] : 0.f;
    Bpack[idx] = (_Float16)v;
}

// ---------------------------------------------------------------------------
// Kernel 1: words[m][h] = sum_k Apack[m][k] * Bpack[h][k] + b_word[h]
// M=4096, Kpad=320, N=64 (50 valid).  f16 WMMA, f32 accumulate.
// 256 blocks x 128 threads; wave w handles N-tile w.  All fragment loads are
// unguarded, aligned vector loads; EXEC is all-1s at every WMMA.
// ---------------------------------------------------------------------------
__global__ __launch_bounds__(128) void words_wmma_kernel(
    const _Float16* __restrict__ Apack, const _Float16* __restrict__ Bpack,
    const float* __restrict__ b_word, float* __restrict__ words)
{
    const int lane  = threadIdx.x & 31;
    const int wave  = threadIdx.x >> 5;
    const int Mbase = blockIdx.x * 16;
    const int Nbase = wave * 16;
    const int half  = lane >> 4;            // 0 or 1
    const int mrow  = Mbase + (lane & 15);  // A row owned by this lane
    const int ncol  = Nbase + (lane & 15);  // B/C/D column owned by this lane

    const _Float16* arow = Apack + mrow * kKpad;
    const _Float16* brow = Bpack + ncol * kKpad;

    v8f acc = {};
    #pragma unroll
    for (int kk = 0; kk < kKpad; kk += 32) {
        // A fragment (16x32 f16): lanes 0-15 hold
        // K=kk+[0..15], lanes 16-31 hold K=kk+[16..31].
        const int ka = kk + half * 16;
        const v8h alo = *(const v8h*)(arow + ka);        // 16B aligned
        const v8h ahi = *(const v8h*)(arow + ka + 8);    // 16B aligned
        const v16h a = __builtin_shufflevector(alo, ahi,
            0, 1, 2, 3, 4, 5, 6, 7, 8, 9, 10, 11, 12, 13, 14, 15);

        // B fragment (32x16 f16): lanes 0-15 hold K=kk+[0..15],
        // lanes 16-31 hold K=kk+[16..31]; column = lane%16.
        const int kb = kk + half * 16;
        const v16h b = *(const v16h*)(brow + kb);        // 32B aligned

        acc = __builtin_amdgcn_wmma_f32_16x16x32_f16(
                  false, a, false, b, (short)0, acc, false, false);
    }

    if (ncol < kH) {
        const float bias = b_word[ncol];
        #pragma unroll
        for (int r = 0; r < 8; ++r) {
            const int m = Mbase + r + half * 8;  // C/D layout: VGPR r -> M=r / 8+r
            words[m * kH + ncol] = acc[r] + bias;
        }
    }
}

// ---------------------------------------------------------------------------
// Kernel 2: conv(k=1,2,3) + time-mean + tanh, averaged -> reps[s][h].
// mean over time of a VALID conv == windowed column sums (prefix/suffix
// corrections of the total), so each conv collapses to one small matvec.
// 64 blocks (one per sentence) x 64 threads.
// ---------------------------------------------------------------------------
__global__ __launch_bounds__(64) void conv_reps_kernel(
    const float* __restrict__ words,
    const float* __restrict__ w1, const float* __restrict__ b1,
    const float* __restrict__ w2, const float* __restrict__ b2,
    const float* __restrict__ w3, const float* __restrict__ b3,
    float* __restrict__ reps)
{
    __shared__ float W10[kH], W20[kH], W21[kH], W30[kH], W31[kH], W32[kH];
    const int s = blockIdx.x;
    const int i = threadIdx.x;

    if (i < kH) {
        const float* base = words + (s * kW) * kH + i;
        float tot = 0.f, p0 = 0.f, p1 = 0.f, l62 = 0.f, l63 = 0.f;
        for (int t = 0; t < kW; ++t) {
            const float v = base[t * kH];
            tot += v;
            if (t == 0)  p0  = v;
            if (t == 1)  p1  = v;
            if (t == 62) l62 = v;
            if (t == 63) l63 = v;
        }
        W10[i] = tot;                    // k=1, tau=0 : t in [0,63]
        W20[i] = tot - l63;              // k=2, tau=0 : t in [0,62]
        W21[i] = tot - p0;               // k=2, tau=1 : t in [1,63]
        W30[i] = tot - l62 - l63;        // k=3, tau=0 : t in [0,61]
        W31[i] = tot - p0 - l63;         // k=3, tau=1 : t in [1,62]
        W32[i] = tot - p0 - p1;          // k=3, tau=2 : t in [2,63]
    }
    __syncthreads();

    const int h = threadIdx.x;
    if (h < kH) {
        float a1 = 0.f, a2 = 0.f, a3 = 0.f;
        for (int c = 0; c < kH; ++c) {
            a1 += w1[h * kH + c] * W10[c];
            a2 += w2[(h * kH + c) * 2 + 0] * W20[c]
                + w2[(h * kH + c) * 2 + 1] * W21[c];
            a3 += w3[(h * kH + c) * 3 + 0] * W30[c]
                + w3[(h * kH + c) * 3 + 1] * W31[c]
                + w3[(h * kH + c) * 3 + 2] * W32[c];
        }
        const float r1 = tanhf(a1 * (1.f / 64.f) + b1[h]);
        const float r2 = tanhf(a2 * (1.f / 63.f) + b2[h]);
        const float r3 = tanhf(a3 * (1.f / 62.f) + b3[h]);
        reps[s * kH + h] = (r1 + r2 + r3) * (1.f / 3.f);
    }
}

// ---------------------------------------------------------------------------
// Kernel 3: forward + backward recurrent scans (parallel halves of one
// block), time-mean of hidden states, output matvec + softmax.
// 1 block x 128 threads (lanes 0-49 = forward, 64-113 = backward).
// ---------------------------------------------------------------------------
__global__ __launch_bounds__(128) void scan_out_kernel(
    const float* __restrict__ reps,
    const float* __restrict__ Wfi, const float* __restrict__ bfi,
    const float* __restrict__ Wff, const float* __restrict__ bff,
    const float* __restrict__ Wfg, const float* __restrict__ bfg,
    const float* __restrict__ Wbi, const float* __restrict__ bbi,
    const float* __restrict__ Wbf, const float* __restrict__ bbf,
    const float* __restrict__ Wbg, const float* __restrict__ bbg,
    const float* __restrict__ Wout, const float* __restrict__ bout,
    float* __restrict__ out)
{
    __shared__ float R[kS * kH];
    __shared__ float hf[kH], hb[kH], nhf[kH], nhb[kH];
    __shared__ float sf[kH], sb[kH];
    __shared__ float gnn[2 * kH];
    __shared__ float logits[kO];

    const int tid = threadIdx.x;
    for (int idx = tid; idx < kS * kH; idx += blockDim.x) R[idx] = reps[idx];
    if (tid < kH)                   { hf[tid] = 0.f;      sf[tid] = 0.f; }
    if (tid >= 64 && tid < 64 + kH) { hb[tid - 64] = 0.f; sb[tid - 64] = 0.f; }
    __syncthreads();

    for (int t = 0; t < kS; ++t) {
        if (tid < kH) {
            const float* x = &R[t * kH];
            float ai = bfi[tid], af = bff[tid], ag = bfg[tid];
            for (int j = 0; j < kH; ++j) {
                const float xj = x[j], hj = hf[j];
                ai += Wfi[tid * 100 + j] * xj + Wfi[tid * 100 + kH + j] * hj;
                af += Wff[tid * 100 + j] * xj + Wff[tid * 100 + kH + j] * hj;
                ag += Wfg[tid * 100 + j] * xj + Wfg[tid * 100 + kH + j] * hj;
            }
            const float it = 1.f / (1.f + expf(-ai));
            const float ft = 1.f / (1.f + expf(-af));
            const float gt = tanhf(ag);
            nhf[tid] = tanhf(it * gt + ft * hf[tid]);
        } else if (tid >= 64 && tid < 64 + kH) {
            const int h = tid - 64;
            const float* x = &R[(kS - 1 - t) * kH];
            float ai = bbi[h], af = bbf[h], ag = bbg[h];
            for (int j = 0; j < kH; ++j) {
                const float xj = x[j], hj = hb[j];
                ai += Wbi[h * 100 + j] * xj + Wbi[h * 100 + kH + j] * hj;
                af += Wbf[h * 100 + j] * xj + Wbf[h * 100 + kH + j] * hj;
                ag += Wbg[h * 100 + j] * xj + Wbg[h * 100 + kH + j] * hj;
            }
            const float it = 1.f / (1.f + expf(-ai));
            const float ft = 1.f / (1.f + expf(-af));
            const float gt = tanhf(ag);
            nhb[h] = tanhf(it * gt + ft * hb[h]);
        }
        __syncthreads();
        if (tid < kH)                   { hf[tid] = nhf[tid]; sf[tid] += nhf[tid]; }
        if (tid >= 64 && tid < 64 + kH) { hb[tid-64] = nhb[tid-64]; sb[tid-64] += nhb[tid-64]; }
        __syncthreads();
    }

    if (tid < kH)                   gnn[tid]           = sf[tid] * (1.f / kS);
    if (tid >= 64 && tid < 64 + kH) gnn[kH + tid - 64] = sb[tid - 64] * (1.f / kS);
    __syncthreads();

    if (tid < kO) {
        float a = bout[tid];
        for (int j = 0; j < 2 * kH; ++j) a += Wout[tid * 2 * kH + j] * gnn[j];
        logits[tid] = a;
    }
    __syncthreads();

    if (tid == 0) {
        float mx = logits[0];
        for (int o = 1; o < kO; ++o) mx = fmaxf(mx, logits[o]);
        float e[kO], sum = 0.f;
        for (int o = 0; o < kO; ++o) { e[o] = expf(logits[o] - mx); sum += e[o]; }
        for (int o = 0; o < kO; ++o) out[o] = e[o] / sum;
    }
}

// ---------------------------------------------------------------------------
extern "C" void kernel_launch(void* const* d_in, const int* in_sizes, int n_in,
                              void* d_out, int out_size, void* d_ws, size_t ws_size,
                              hipStream_t stream)
{
    const int*   doc    = (const int*)  d_in[0];
    const float* emb    = (const float*)d_in[1];
    const float* W_word = (const float*)d_in[2];
    const float* b_word = (const float*)d_in[3];
    const float* w1     = (const float*)d_in[4];
    const float* b1     = (const float*)d_in[5];
    const float* w2     = (const float*)d_in[6];
    const float* b2     = (const float*)d_in[7];
    const float* w3     = (const float*)d_in[8];
    const float* b3     = (const float*)d_in[9];
    const float* Wfi    = (const float*)d_in[10];
    const float* bfi    = (const float*)d_in[11];
    const float* Wff    = (const float*)d_in[12];
    const float* bff    = (const float*)d_in[13];
    const float* Wfg    = (const float*)d_in[14];
    const float* bfg    = (const float*)d_in[15];
    const float* Wbi    = (const float*)d_in[16];
    const float* bbi    = (const float*)d_in[17];
    const float* Wbf    = (const float*)d_in[18];
    const float* bbf    = (const float*)d_in[19];
    const float* Wbg    = (const float*)d_in[20];
    const float* bbg    = (const float*)d_in[21];
    const float* Wout   = (const float*)d_in[22];
    const float* bout   = (const float*)d_in[23];

    // Workspace layout (bytes):
    //   Apack : kM * kKpad * 2      = 2,621,440
    //   Bpack : kNpad * kKpad * 2   = 40,960
    //   words : kM * kH * 4         = 819,200
    //   reps  : kS * kH * 4         = 12,800
    char* ws = (char*)d_ws;
    _Float16* Apack = (_Float16*)ws;
    _Float16* Bpack = (_Float16*)(ws + (size_t)kM * kKpad * 2);
    float*    words = (float*)   (ws + (size_t)kM * kKpad * 2 + (size_t)kNpad * kKpad * 2);
    float*    reps  = (float*)   ((char*)words + (size_t)kM * kH * 4);

    pack_A_kernel<<<kM, 160, 0, stream>>>(doc, emb, (unsigned*)Apack);
    pack_B_kernel<<<(kNpad * kKpad + 255) / 256, 256, 0, stream>>>(W_word, Bpack);
    words_wmma_kernel<<<kM / 16, 128, 0, stream>>>(Apack, Bpack, b_word, words);
    conv_reps_kernel<<<kS, 64, 0, stream>>>(words, w1, b1, w2, b2, w3, b3, reps);
    scan_out_kernel<<<1, 128, 0, stream>>>(reps,
        Wfi, bfi, Wff, bff, Wfg, bfg,
        Wbi, bbi, Wbf, bbf, Wbg, bbg,
        Wout, bout, (float*)d_out);
}
